// EffDistinctStackNodesOnAffinityValueAttention_21079699489153
// MI455X (gfx1250) — matrix operand order repaired and checked by reading, not compiled, not run on hardware
//
#include <hip/hip_runtime.h>

typedef __attribute__((ext_vector_type(16))) _Float16 v16h;
typedef __attribute__((ext_vector_type(8)))  float    v8f;

#define B_   4
#define H_   8
#define TK_  512
#define NK_  511
#define L_   1023
#define D_   512
#define ROWS 4092     /* B_*L_ */
#define LP   1024

#define QLEAF 0x40000000u
#define QNODE 0x20000000u
#define PADF  0x10000000u

struct HalfFrag { uint4 a, b; };

__device__ inline v8f wmma16(v16h a, v16h b, v8f c) {
  // emits v_wmma_f32_16x16x32_f16
  return __builtin_amdgcn_wmma_f32_16x16x32_f16(false, a, false, b, (short)0, c,
                                                false, false);
}

__device__ inline unsigned short f2h(float f) {
  _Float16 h = (_Float16)f;
  return __builtin_bit_cast(unsigned short, h);
}

// A-fragment (16x32, 16-bit): lane(hb,m): elems K = koff+16hb..+15 (contiguous)
__device__ inline v16h ldfragA(const unsigned short* p, int koff, int hb) {
  HalfFrag s;
  s.a = *(const uint4*)(p + koff + 16 * hb);
  s.b = *(const uint4*)(p + koff + 16 * hb + 8);
  return __builtin_bit_cast(v16h, s);
}
// B-fragment (32x16, 16-bit): lane(hb,n): elems K = koff+16hb..+15 (contiguous)
__device__ inline v16h ldfragB(const unsigned short* p, int koff, int hb) {
  HalfFrag s;
  s.a = *(const uint4*)(p + koff + 16 * hb);
  s.b = *(const uint4*)(p + koff + 16 * hb + 8);
  return __builtin_bit_cast(v16h, s);
}

__device__ inline bool is_masked(unsigned qp, unsigned kp, int q, int t) {
  if (!(qp & (QLEAF | QNODE))) return true;           // pad query row
  if (qp & QLEAF) {                                   // leaf query: unpadded leaf keys only
    return !((kp & QLEAF) && !(kp & PADF));
  }
  // node query
  if (kp & QLEAF) {                                   // leaf key: inside span, no pads
    int lo = (int)(qp & 1023u), hi = (int)((qp >> 10) & 1023u);
    return (t < lo) || (t > hi) || ((qp & PADF) != 0u) || ((kp & PADF) != 0u);
  }
  if (kp & QNODE) {                                   // node key: tril + span overlap
    int qlo = (int)(qp & 1023u), qhi = (int)((qp >> 10) & 1023u);
    int klo = (int)(kp & 1023u), khi = (int)((kp >> 10) & 1023u);
    int mlo = qlo > klo ? qlo : klo;
    int mhi = qhi < khi ? qhi : khi;
    bool keep = (t <= q) && (mlo <= mhi) && !(kp & PADF);
    return !keep;
  }
  return true;                                        // pad key col
}

// 16-byte global -> LDS async copy (CDNA5 TDM-lite path, tracked by ASYNCcnt)
__device__ inline void async_cp16(unsigned lds_addr, const char* gaddr) {
  asm volatile("global_load_async_to_lds_b128 %0, %1, off"
               :: "v"(lds_addr), "v"(gaddr) : "memory");
}

// ---------------- conversion kernels ----------------
__global__ void cvt_x_kernel(const float* __restrict__ leaves,
                             const float* __restrict__ nodes,
                             unsigned short* __restrict__ Xh) {
  int idx = blockIdx.x * 256 + threadIdx.x;
  if (idx >= ROWS * D_) return;
  int row = idx >> 9, col = idx & 511;
  int b = row / L_, l = row - b * L_;
  float v = (l < TK_) ? leaves[(size_t)(b * TK_ + l) * D_ + col]
                      : nodes[(size_t)(b * NK_ + (l - TK_)) * D_ + col];
  Xh[idx] = f2h(v);
}

__global__ void cvt_w_kernel(const float* __restrict__ W,
                             unsigned short* __restrict__ Wh) {
  int idx = blockIdx.x * 256 + threadIdx.x;
  if (idx < D_ * D_) Wh[idx] = f2h(W[idx]);
}

__global__ void pad_zero_kernel(unsigned short* __restrict__ Qh,
                                unsigned short* __restrict__ Kh,
                                unsigned short* __restrict__ Vt) {
  int idx = blockIdx.x * 256 + threadIdx.x;
  if (idx >= B_ * H_ * 64) return;
  int bh = idx >> 6, d = idx & 63;
  Qh[((size_t)(bh * LP + (LP - 1))) * 64 + d] = 0;
  Kh[((size_t)(bh * LP + (LP - 1))) * 64 + d] = 0;
  Vt[((size_t)(bh * 64 + d)) * LP + (LP - 1)] = 0;
}

// ---------------- WMMA GEMM: C(16x64/wave) = A(f16) * W(f16)^T + bias ----------
// MODE 0: Q (scale 1/8, store (B,H,LP,64) f16)   MODE 1: K (same layout)
// MODE 2: V (store transposed (B,H,64,LP) f16)   MODE 3: out proj (f32 rows x 512)
template <int MODE>
__global__ __launch_bounds__(32) void gemm16_kernel(
    const unsigned short* __restrict__ A, const unsigned short* __restrict__ W,
    const float* __restrict__ bias, void* __restrict__ dst) {
  const int lane = threadIdx.x;
  const int hb = lane >> 4, ln = lane & 15;
  const int j0 = blockIdx.x * 64, r0 = blockIdx.y * 16;

  int arow = r0 + ln;
  if (arow > ROWS - 1) arow = ROWS - 1;
  const unsigned short* Ar = A + (size_t)arow * D_;
  const unsigned short* Wr0 = W + (size_t)(j0 + ln) * D_;
  const unsigned short* Wr1 = Wr0 + (size_t)16 * D_;
  const unsigned short* Wr2 = Wr0 + (size_t)32 * D_;
  const unsigned short* Wr3 = Wr0 + (size_t)48 * D_;

  v8f acc0 = {}, acc1 = {}, acc2 = {}, acc3 = {};
#pragma unroll 4
  for (int kc = 0; kc < 16; ++kc) {
    v16h af = ldfragA(Ar, kc * 32, hb);       // reused across 4 column tiles
    acc0 = wmma16(af, ldfragB(Wr0, kc * 32, hb), acc0);
    acc1 = wmma16(af, ldfragB(Wr1, kc * 32, hb), acc1);
    acc2 = wmma16(af, ldfragB(Wr2, kc * 32, hb), acc2);
    acc3 = wmma16(af, ldfragB(Wr3, kc * 32, hb), acc3);
  }

  v8f accs[4] = {acc0, acc1, acc2, acc3};
  const float scale = (MODE == 0) ? 0.125f : 1.0f;  // HD^-0.5 folded into Q

#pragma unroll
  for (int c = 0; c < 4; ++c) {
    const int j = j0 + 16 * c + ln;
    const float bval = bias[j];
#pragma unroll
    for (int r = 0; r < 8; ++r) {
      int row = r0 + 8 * hb + r;
      if (row >= ROWS) continue;
      float v = (accs[c][r] + bval) * scale;
      int b = row / L_;
      int l = row - b * L_;
      if (MODE == 0 || MODE == 1) {
        unsigned short* o = (unsigned short*)dst;
        o[((size_t)((b * H_ + (j >> 6)) * LP + l)) * 64 + (j & 63)] = f2h(v);
      } else if (MODE == 2) {
        unsigned short* o = (unsigned short*)dst;
        o[((size_t)((b * H_ + (j >> 6)) * 64 + (j & 63))) * LP + l] = f2h(v);
      } else {
        float* o = (float*)dst;
        o[(size_t)row * D_ + j] = v;
      }
    }
  }
}

// ---------------- flash attention, async double-buffered K/V ----------------
__global__ __launch_bounds__(32) void attn_kernel(
    const unsigned short* __restrict__ Qh, const unsigned short* __restrict__ Kh,
    const unsigned short* __restrict__ Vt, const int* __restrict__ nidx,
    const unsigned char* __restrict__ key_pad,
    const unsigned char* __restrict__ node_pad,
    unsigned short* __restrict__ Ao) {
  __shared__ __align__(16) unsigned short Kbuf[2][32 * 64];  // 2 x 4 KB
  __shared__ __align__(16) unsigned short Vbuf[2][64 * 32];  // 2 x 4 KB
  __shared__ __align__(16) unsigned short Pbuf[16 * 32];     // 1 KB
  __shared__ unsigned Mtab[LP];                              // packed mask words, 4 KB

  const int lane = threadIdx.x;
  const int hb = lane >> 4, ln = lane & 15;
  const int q0 = blockIdx.x * 16;  // 64 query tiles
  const int bh = blockIdx.y;       // 32 (b,h)
  const int b = bh >> 3, h = bh & 7;

  // ---- build packed mask table for all 1024 positions (shared q/k encoding) ----
  for (int tt = 0; tt < 32; ++tt) {
    int t = tt * 32 + lane;
    unsigned w = 0;
    if (t < TK_) {
      w = QLEAF;
      if (key_pad[b * TK_ + t]) w |= PADF;
    } else if (t < L_) {
      int j = t - TK_;
      w = QNODE | (unsigned)nidx[(size_t)(bh * NK_ + j) * 2 + 0] |
          ((unsigned)nidx[(size_t)(bh * NK_ + j) * 2 + 1] << 10);
      if (node_pad[b * NK_ + j]) w |= PADF;
    }
    Mtab[t] = w;
  }
  __syncthreads();

  // ---- Q fragments for both K-chunks (dims 0..31, 32..63) ----
  const unsigned short* Qrow = Qh + (size_t)(bh * LP + q0 + ln) * 64;
  v16h Aq0 = ldfragA(Qrow, 0, hb);
  v16h Aq1 = ldfragA(Qrow, 32, hb);

  unsigned qinfo[8];
  int qrow[8];
#pragma unroll
  for (int r = 0; r < 8; ++r) {
    qrow[r] = q0 + r + 8 * hb;
    qinfo[r] = Mtab[qrow[r]];
  }

  const char* KbaseG = (const char*)(Kh + (size_t)bh * LP * 64);  // row t: +t*128 B
  const char* VbaseG = (const char*)(Vt + (size_t)bh * 64 * LP);  // row d: +d*2048 B
  const unsigned kl0 = (unsigned)(uintptr_t)(void*)&Kbuf[0][0];
  const unsigned kl1 = (unsigned)(uintptr_t)(void*)&Kbuf[1][0];
  const unsigned vl0 = (unsigned)(uintptr_t)(void*)&Vbuf[0][0];
  const unsigned vl1 = (unsigned)(uintptr_t)(void*)&Vbuf[1][0];

  // stage K/V tile kt into buffer bsel (16 async b128 per lane)
  auto stage = [&](int kt, int bsel) {
    const char* kg = KbaseG + (size_t)kt * 32 * 128;  // 4 KB contiguous
    unsigned kl = bsel ? kl1 : kl0;
    unsigned vl = bsel ? vl1 : vl0;
#pragma unroll
    for (int i = 0; i < 8; ++i) {
      unsigned off = (unsigned)(lane * 16 + i * 512);
      async_cp16(kl + off, kg + off);
    }
    const char* vg = VbaseG + (size_t)kt * 64;        // col offset t0*2 bytes
#pragma unroll
    for (int i = 0; i < 8; ++i) {
      unsigned c = (unsigned)(lane + 32 * i);
      async_cp16(vl + c * 16, vg + (size_t)(c >> 2) * 2048 + (c & 3) * 16);
    }
  };

  float mrun[8], lsum[8];
  v8f Oc[4];
#pragma unroll
  for (int r = 0; r < 8; ++r) { mrun[r] = -1e30f; lsum[r] = 0.0f; }
#pragma unroll
  for (int c = 0; c < 4; ++c) { v8f z = {}; Oc[c] = z; }

  stage(0, 0);

  for (int kt = 0; kt < 32; ++kt) {  // 32 keys per step
    const int t0 = kt * 32;
    const int buf = kt & 1;

    if (kt < 31) {
      stage(kt + 1, buf ^ 1);
      asm volatile("s_wait_asynccnt 0x10" ::: "memory");  // tile kt landed
    } else {
      asm volatile("s_wait_asynccnt 0x0" ::: "memory");
    }

    unsigned kp0 = Mtab[t0 + ln];
    unsigned kp1 = Mtab[t0 + 16 + ln];
    const int te0 = t0 + ln, te1 = t0 + 16 + ln;

    // S = Q K^T  (two 16x16 tiles, two K-chunks each) from LDS
    const unsigned short* Kr0 = &Kbuf[buf][0] + ln * 64;
    const unsigned short* Kr1 = Kr0 + 16 * 64;
    v8f c0 = {}, c1 = {};
    c0 = wmma16(Aq0, ldfragB(Kr0, 0, hb), c0);
    c0 = wmma16(Aq1, ldfragB(Kr0, 32, hb), c0);
    c1 = wmma16(Aq0, ldfragB(Kr1, 0, hb), c1);
    c1 = wmma16(Aq1, ldfragB(Kr1, 32, hb), c1);

    // mask + online softmax, stage P into LDS as f16
#pragma unroll
    for (int r = 0; r < 8; ++r) {
      float s0 = c0[r], s1 = c1[r];
      if (is_masked(qinfo[r], kp0, qrow[r], te0)) s0 = -1e30f;
      if (is_masked(qinfo[r], kp1, qrow[r], te1)) s1 = -1e30f;
      float mx = fmaxf(s0, s1);
#pragma unroll
      for (int off = 1; off < 16; off <<= 1) mx = fmaxf(mx, __shfl_xor(mx, off, 32));
      float mnew = fmaxf(mrun[r], mx);
      float alpha = __expf(mrun[r] - mnew);
      float p0 = __expf(s0 - mnew);
      float p1 = __expf(s1 - mnew);
      float ps = p0 + p1;
#pragma unroll
      for (int off = 1; off < 16; off <<= 1) ps += __shfl_xor(ps, off, 32);
      lsum[r] = lsum[r] * alpha + ps;
      mrun[r] = mnew;
#pragma unroll
      for (int c = 0; c < 4; ++c) Oc[c][r] *= alpha;
      int m = r + 8 * hb;
      Pbuf[m * 32 + ln] = f2h(p0);
      Pbuf[m * 32 + 16 + ln] = f2h(p1);
    }
    __syncthreads();

    // P (16x32) x V (32x64): 4 WMMAs over dim-chunks, V from LDS
    v16h Ap = ldfragA(Pbuf + ln * 32, 0, hb);
    const unsigned short* Vb = &Vbuf[buf][0] + ln * 32 + 16 * hb;
#pragma unroll
    for (int c = 0; c < 4; ++c) {
      HalfFrag s;
      s.a = *(const uint4*)(Vb + c * 16 * 32);
      s.b = *(const uint4*)(Vb + c * 16 * 32 + 8);
      v16h Bv = __builtin_bit_cast(v16h, s);
      Oc[c] = wmma16(Ap, Bv, Oc[c]);
    }
    __syncthreads();
  }

  // normalize and store (B,L,D) f16, D-index = h*64+d
#pragma unroll
  for (int r = 0; r < 8; ++r) {
    int l = q0 + r + 8 * hb;
    if (l >= L_) continue;
    float inv = 1.0f / lsum[r];
#pragma unroll
    for (int c = 0; c < 4; ++c) {
      Ao[(size_t)(b * L_ + l) * D_ + h * 64 + c * 16 + ln] = f2h(Oc[c][r] * inv);
    }
  }
}

// ---------------- host launch ----------------
extern "C" void kernel_launch(void* const* d_in, const int* in_sizes, int n_in,
                              void* d_out, int out_size, void* d_ws, size_t ws_size,
                              hipStream_t stream) {
  const float* leaves = (const float*)d_in[0];
  const float* nodes = (const float*)d_in[1];
  const float* Wq = (const float*)d_in[2];
  const float* Wk = (const float*)d_in[3];
  const float* Wv = (const float*)d_in[4];
  const float* Wo = (const float*)d_in[5];
  const float* bq = (const float*)d_in[6];
  const float* bk = (const float*)d_in[7];
  const float* bv = (const float*)d_in[8];
  const float* bo = (const float*)d_in[9];
  const int* nidx = (const int*)d_in[10];
  const unsigned char* key_pad = (const unsigned char*)d_in[11];
  const unsigned char* node_pad = (const unsigned char*)d_in[12];

  char* ws = (char*)d_ws;
  unsigned short* Xh  = (unsigned short*)(ws + 0);          // 4096x512 f16  (4 MB)
  unsigned short* Wqh = (unsigned short*)(ws + 4194304);    // 512x512 f16
  unsigned short* Wkh = (unsigned short*)(ws + 4718592);
  unsigned short* Wvh = (unsigned short*)(ws + 5242880);
  unsigned short* Woh = (unsigned short*)(ws + 5767168);
  unsigned short* Qh  = (unsigned short*)(ws + 6291456);    // (B,H,1024,64) f16
  unsigned short* Kh  = (unsigned short*)(ws + 10485760);   // (B,H,1024,64) f16
  unsigned short* Vt  = (unsigned short*)(ws + 14680064);   // (B,H,64,1024) f16
  unsigned short* Ao  = (unsigned short*)(ws + 18874368);   // 4096x512 f16

  cvt_x_kernel<<<(ROWS * D_ + 255) / 256, 256, 0, stream>>>(leaves, nodes, Xh);
  cvt_w_kernel<<<(D_ * D_ + 255) / 256, 256, 0, stream>>>(Wq, Wqh);
  cvt_w_kernel<<<(D_ * D_ + 255) / 256, 256, 0, stream>>>(Wk, Wkh);
  cvt_w_kernel<<<(D_ * D_ + 255) / 256, 256, 0, stream>>>(Wv, Wvh);
  cvt_w_kernel<<<(D_ * D_ + 255) / 256, 256, 0, stream>>>(Wo, Woh);
  pad_zero_kernel<<<8, 256, 0, stream>>>(Qh, Kh, Vt);

  dim3 ggrid(D_ / 64, 4096 / 16);  // 8 x 256 tiles of 16x64
  gemm16_kernel<0><<<ggrid, 32, 0, stream>>>(Xh, Wqh, bq, (void*)Qh);
  gemm16_kernel<1><<<ggrid, 32, 0, stream>>>(Xh, Wkh, bk, (void*)Kh);
  gemm16_kernel<2><<<ggrid, 32, 0, stream>>>(Xh, Wvh, bv, (void*)Vt);

  attn_kernel<<<dim3(LP / 16, B_ * H_), 32, 0, stream>>>(Qh, Kh, Vt, nidx,
                                                         key_pad, node_pad, Ao);

  gemm16_kernel<3><<<ggrid, 32, 0, stream>>>(Ao, Woh, bo, d_out);
}
